// myModel_67087389163867
// MI455X (gfx1250) — matrix operand order repaired and checked by reading, not compiled, not run on hardware
//
#include <hip/hip_runtime.h>
#include <hip/hip_bf16.h>

// ---------------------------------------------------------------------------
// RouteNet-style critic: T=4 message-passing steps over 1.6M edge-pairs,
// D=64 link state, GRU update, readout MLP 64->256->256->1.
//
// Message MLP [P,128]x[128,64] (~26 GFLOP/step)  -> bf16 WMMA, fused scatter.
// GRU GEMMs  2x [E,64]x[64,192] (~10 GFLOP/step) -> bf16 WMMA, 32 edges/block.
// Reduction + readout MLP: bandwidth-trivial, plain VALU.
// ---------------------------------------------------------------------------

typedef __attribute__((ext_vector_type(16))) __bf16 v16bf;
typedef __attribute__((ext_vector_type(8)))  float  v8f;

#define LSD 64     // link state dim D
#define TWOD 128   // 2*D (concat)
#define RU 256     // readout units

__device__ __forceinline__ unsigned short f32_to_bf16_rne(float f) {
    unsigned u = __float_as_uint(f);
    unsigned r = u + 0x7FFFu + ((u >> 16) & 1u);
    return (unsigned short)(r >> 16);
}

__device__ __forceinline__ unsigned pack_bf16x2(float lo, float hi) {
    return (unsigned)f32_to_bf16_rne(lo) | ((unsigned)f32_to_bf16_rne(hi) << 16);
}

// Branchless SELU: second term is exactly 0 for x >= 0.
__device__ __forceinline__ float selu_f(float x) {
    const float sc   = 1.0507009873554805f;
    const float scal = 1.7580993408473766f;   // sc * alpha
    return sc * fmaxf(x, 0.f) + scal * (__expf(fminf(x, 0.f)) - 1.f);
}

__device__ __forceinline__ float sigmoid_f(float x) {
    return 1.f / (1.f + __expf(-x));
}

// -------------------------------- utility ----------------------------------

__global__ __launch_bounds__(256) void copy_f32_kernel(float* __restrict__ dst,
                                                       const float* __restrict__ src,
                                                       long n) {
    long i = (long)blockIdx.x * blockDim.x + threadIdx.x;
    long stride = (long)gridDim.x * blockDim.x;
    for (; i < n; i += stride) dst[i] = src[i];
}

__global__ __launch_bounds__(256) void zero_f32_kernel(float* __restrict__ dst, long n) {
    long i = (long)blockIdx.x * blockDim.x + threadIdx.x;
    long stride = (long)gridDim.x * blockDim.x;
    for (; i < n; i += stride) dst[i] = 0.f;
}

// ------------------ f32 [K,N] -> bf16 per-lane B fragments ------------------
// Fragment order: tile = ntile*KSTEPS + ks; per (tile, lane, j, half):
//   n = ntile*16 + (lane & 15), g = lane>>4
//   k = ks*32 + g*8 + 2*j + half + (j>=4 ? 8 : 0)   (dense 16-bit B 32x16 lane layout)
// flat dst index = ((tile*32 + lane)*16 + j*2 + half)  -> 32B/lane contiguous.
__global__ __launch_bounds__(256) void prep_frag_kernel(const float* __restrict__ src,
                                                        unsigned short* __restrict__ dst,
                                                        int Ktot, int Ntot) {
    int ksteps = Ktot >> 5;
    int total  = (Ktot >> 5) * (Ntot >> 4) * 512;
    for (int t = blockIdx.x * 256 + threadIdx.x; t < total; t += gridDim.x * 256) {
        int half = t & 1;
        int j    = (t >> 1) & 7;
        int lane = (t >> 4) & 31;
        int tile = t >> 9;
        int ks    = tile % ksteps;
        int ntile = tile / ksteps;
        int n = ntile * 16 + (lane & 15);
        int k = ks * 32 + (lane >> 4) * 8 + 2 * j + half + ((j >= 4) ? 8 : 0);
        dst[t] = f32_to_bf16_rne(src[k * Ntot + n]);
    }
}

// --------------------- fused gather + msg-MLP + scatter ---------------------
// Block = 256 threads = 8 wave32: wave w -> rowtile = w>>2 (0..1), ntile = w&3.
// Block covers 32 pairs x 64 output cols. K = 128 via 4 bf16 WMMA steps.
__global__ __launch_bounds__(256) void msg_scatter_kernel(
        const float* __restrict__ ls,            // [E,64] current link state
        const int*   __restrict__ first,
        const int*   __restrict__ second,
        const unsigned short* __restrict__ wfrag,// bf16 B fragments (W_msg)
        const float* __restrict__ b_msg,         // [64]
        float* __restrict__ agg,                 // [E,64] accumulator
        long P) {
    __shared__ unsigned As[32 * 64];             // 32 pair rows x 128 bf16 (packed)
    __shared__ int sidx[32];
    __shared__ int fidx[32];

    int t = threadIdx.x;
    long pairbase = (long)blockIdx.x * 32;
    bool full = (pairbase + 32) <= P;            // uniform per block

    if (t < 32) {
        long p = pairbase + t;
        int f = 0, s = 0;
        if (full || p < P) { f = first[p]; s = second[p]; }
        fidx[t] = f;
        sidx[t] = s;
    }
    __syncthreads();

    // Stage A: concat(ls[first], ls[second]) rows -> packed bf16 pairs in LDS.
    #pragma unroll
    for (int i = 0; i < 8; ++i) {
        int idx = i * 256 + t;                   // 0..2047 (uint slots)
        int row = idx >> 6;                      // 64 uints per row
        int c2  = idx & 63;
        int col = c2 * 2;                        // bf16 col, even
        int src = (col < LSD) ? fidx[row] : sidx[row];
        float2 v = *(const float2*)(ls + (long)src * LSD + (col & (LSD - 1)));
        As[row * 64 + c2] = pack_bf16x2(v.x, v.y);
    }
    __syncthreads();

    int wave    = t >> 5;
    int lane    = t & 31;
    int rowtile = wave >> 2;                     // 0..1
    int ntile   = wave & 3;                      // 0..3
    int g       = lane >> 4;                     // lane group
    int lrow    = lane & 15;

    v8f c = {0.f, 0.f, 0.f, 0.f, 0.f, 0.f, 0.f, 0.f};

    const unsigned* arow_u = As + (rowtile * 16 + lrow) * 64; // packed pairs
    const v16bf*    bsrc   = (const v16bf*)wfrag;

    #pragma unroll
    for (int ks = 0; ks < 4; ++ks) {
        // A fragment: 16-bit A 16x32 layout.
        // lanes 0-15: VGPR0-3 K=0..7, VGPR4-7 K=16..23; lanes 16-31: +8.
        union { unsigned u[8]; v16bf v; } af;
        int kbase = ks * 32 + g * 8;
        #pragma unroll
        for (int j = 0; j < 4; ++j) af.u[j]     = arow_u[(kbase + 2 * j) >> 1];
        #pragma unroll
        for (int j = 0; j < 4; ++j) af.u[4 + j] = arow_u[(kbase + 16 + 2 * j) >> 1];

        v16bf bf = bsrc[(ntile * 4 + ks) * 32 + lane];

        c = __builtin_amdgcn_wmma_f32_16x16x32_bf16(
                false, af.v, false, bf, (short)0, c, false, false);
    }

    // Epilogue: bias + SELU (branchless) + segment_sum scatter (atomic f32).
    int   n    = ntile * 16 + lrow;
    float bias = b_msg[n];
    if (full) {
        #pragma unroll
        for (int i = 0; i < 8; ++i) {
            int m = i + 8 * g;                   // row within 16-tile (C layout)
            float x = selu_f(c[i] + bias);
            atomicAdd(&agg[(long)sidx[rowtile * 16 + m] * LSD + n], x);
        }
    } else {
        #pragma unroll
        for (int i = 0; i < 8; ++i) {
            int  m = i + 8 * g;
            long p = pairbase + rowtile * 16 + m;
            if (p < P) {
                float x = selu_f(c[i] + bias);
                atomicAdd(&agg[(long)sidx[rowtile * 16 + m] * LSD + n], x);
            }
        }
    }
}

// --------------------------- GRU update via WMMA ----------------------------
// 32 edges per 256-thread block. 8 waves cover 48 16x16 tiles:
//   tiles 0..23  -> mx = x @ kernel   (rowtile 0..1 x ntile 0..11)
//   tiles 24..47 -> mi = h @ rkernel
// Each tile: 2 WMMA k-steps (K=64). Results (+bias) land in LDS, then a
// branch-free elementwise pass computes the gates.
__global__ __launch_bounds__(256) void gru_wmma_kernel(
        const float* __restrict__ agg,            // x  [E,64]
        float* __restrict__ h,                    // in/out [E,64]
        const unsigned short* __restrict__ fgk,   // bf16 frags of kernel  [64,192]
        const unsigned short* __restrict__ fgrk,  // bf16 frags of rkernel [64,192]
        const float* __restrict__ bias,           // [2,192]
        long E) {
    __shared__ unsigned Ax[32 * 32];              // x rows, packed bf16 (4 KB)
    __shared__ unsigned Ah[32 * 32];              // h rows, packed bf16 (4 KB)
    __shared__ float Mx[32 * 192];                // mx + bias           (24 KB)
    __shared__ float Mi[32 * 192];                // mi + bias           (24 KB)

    int t = threadIdx.x;
    long ebase = (long)blockIdx.x * 32;
    bool full = (ebase + 32) <= E;                // uniform per block

    // Stage x and h as packed bf16 (coalesced float2 loads).
    #pragma unroll
    for (int i = 0; i < 4; ++i) {
        int idx = i * 256 + t;                    // 0..1023 (uint slots)
        int row = idx >> 5;                       // 32 uints per row
        int c2  = idx & 31;
        long e = ebase + row;
        float2 xv = make_float2(0.f, 0.f), hv = make_float2(0.f, 0.f);
        if (full || e < E) {
            xv = *(const float2*)(agg + e * LSD + c2 * 2);
            hv = *(const float2*)(h   + e * LSD + c2 * 2);
        }
        Ax[row * 32 + c2] = pack_bf16x2(xv.x, xv.y);
        Ah[row * 32 + c2] = pack_bf16x2(hv.x, hv.y);
    }
    __syncthreads();

    int wave = t >> 5, lane = t & 31, g = lane >> 4, lrow = lane & 15;

    for (int iter = 0; iter < 6; ++iter) {
        int tile = iter * 8 + wave;               // 0..47
        int mat  = (tile >= 24) ? 1 : 0;          // 0: mx, 1: mi
        int tl   = mat ? (tile - 24) : tile;      // 0..23
        int rt   = tl / 12;                       // row tile 0..1
        int nt   = tl % 12;                       // n tile 0..11

        const unsigned* arow_u = (mat ? Ah : Ax) + (rt * 16 + lrow) * 32;
        const unsigned short* Bsrc = mat ? fgrk : fgk;

        v8f c = {0.f, 0.f, 0.f, 0.f, 0.f, 0.f, 0.f, 0.f};
        #pragma unroll
        for (int ks = 0; ks < 2; ++ks) {
            union { unsigned u[8]; v16bf v; } af;
            int kbase = ks * 32 + g * 8;
            #pragma unroll
            for (int j = 0; j < 4; ++j) af.u[j]     = arow_u[(kbase + 2 * j) >> 1];
            #pragma unroll
            for (int j = 0; j < 4; ++j) af.u[4 + j] = arow_u[(kbase + 16 + 2 * j) >> 1];

            v16bf bf = ((const v16bf*)Bsrc)[(nt * 2 + ks) * 32 + lane];

            c = __builtin_amdgcn_wmma_f32_16x16x32_bf16(
                    false, af.v, false, bf, (short)0, c, false, false);
        }

        int   n  = nt * 16 + lrow;
        float bn = bias[mat * 192 + n];
        float* Mdst = mat ? Mi : Mx;
        #pragma unroll
        for (int i = 0; i < 8; ++i) {
            int row = rt * 16 + i + 8 * g;        // C-tile row layout
            Mdst[row * 192 + n] = c[i] + bn;
        }
    }
    __syncthreads();

    // Elementwise gates (branch-free), 2 cols per thread, in-place h update.
    #pragma unroll
    for (int i = 0; i < 4; ++i) {
        int idx = i * 256 + t;                    // 0..1023
        int row = idx >> 5;
        int j   = (idx & 31) * 2;
        long e = ebase + row;
        if (full || e < E) {
            float2 hv = *(const float2*)(h + e * LSD + j);
            const float* mx = Mx + row * 192;
            const float* mi = Mi + row * 192;
            float2 out;
            {
                float z  = sigmoid_f(mx[j] + mi[j]);
                float r  = sigmoid_f(mx[64 + j] + mi[64 + j]);
                float hh = tanhf(mx[128 + j] + r * mi[128 + j]);
                out.x = z * hv.x + (1.f - z) * hh;
            }
            {
                float z  = sigmoid_f(mx[j + 1] + mi[j + 1]);
                float r  = sigmoid_f(mx[65 + j] + mi[65 + j]);
                float hh = tanhf(mx[129 + j] + r * mi[129 + j]);
                out.y = z * hv.y + (1.f - z) * hh;
            }
            *(float2*)(h + e * LSD + j) = out;
        }
    }
}

// ------------------------- sum over edges -> g[64] --------------------------
__global__ __launch_bounds__(256) void reduce_kernel(const float* __restrict__ h,
                                                     float* __restrict__ gsum,
                                                     long E) {
    __shared__ float red[256];
    int t   = threadIdx.x;
    int col = t & 63;
    int grp = t >> 6;
    float acc = 0.f;
    long stride = (long)gridDim.x * 4;
    for (long e = (long)blockIdx.x * 4 + grp; e < E; e += stride)
        acc += h[e * LSD + col];
    red[t] = acc;
    __syncthreads();
    if (t < 64)
        atomicAdd(&gsum[t], red[t] + red[t + 64] + red[t + 128] + red[t + 192]);
}

// ------------------------------ readout MLP ---------------------------------
__global__ __launch_bounds__(256) void mlp_kernel(
        const float* __restrict__ gsum,
        const float* __restrict__ W1, const float* __restrict__ b1,
        const float* __restrict__ W2, const float* __restrict__ b2,
        const float* __restrict__ W3, const float* __restrict__ b3,
        float* __restrict__ out) {
    __shared__ float g[LSD];
    __shared__ float h1[RU];
    __shared__ float h2[RU];
    int t = threadIdx.x;
    if (t < LSD) g[t] = gsum[t];
    __syncthreads();

    float a = b1[t];
    #pragma unroll 8
    for (int k = 0; k < LSD; ++k) a = __builtin_fmaf(g[k], W1[k * RU + t], a);
    h1[t] = selu_f(a);
    __syncthreads();

    float c = b2[t];
    #pragma unroll 8
    for (int k = 0; k < RU; ++k) c = __builtin_fmaf(h1[k], W2[k * RU + t], c);
    h2[t] = selu_f(c) * W3[t];
    __syncthreads();

    for (int s = 128; s > 0; s >>= 1) {
        if (t < s) h2[t] += h2[t + s];
        __syncthreads();
    }
    if (t == 0) out[0] = h2[0] + b3[0];
}

// ---------------------------------------------------------------------------

extern "C" void kernel_launch(void* const* d_in, const int* in_sizes, int n_in,
                              void* d_out, int out_size, void* d_ws, size_t ws_size,
                              hipStream_t stream) {
    const float* link_state = (const float*)d_in[0];
    const int*   first      = (const int*)d_in[1];
    const int*   second     = (const int*)d_in[2];
    //            d_in[3] = num_edges scalar (derived from in_sizes instead)
    const float* W_msg      = (const float*)d_in[4];
    const float* b_msg      = (const float*)d_in[5];
    const float* gk         = (const float*)d_in[6];
    const float* grk        = (const float*)d_in[7];
    const float* gb         = (const float*)d_in[8];
    const float* W1         = (const float*)d_in[9];
    const float* b1         = (const float*)d_in[10];
    const float* W2         = (const float*)d_in[11];
    const float* b2         = (const float*)d_in[12];
    const float* W3         = (const float*)d_in[13];
    const float* b3         = (const float*)d_in[14];

    long E = (long)in_sizes[0] / LSD;
    long P = (long)in_sizes[1];

    // Workspace layout
    size_t lsz = (size_t)E * LSD * sizeof(float);
    char* ws = (char*)d_ws;
    float*          ls        = (float*)ws;                          // [E,64]
    float*          agg       = (float*)(ws + lsz);                  // [E,64]
    unsigned short* wfrag_msg = (unsigned short*)(ws + 2 * lsz);     // 16 KB
    unsigned short* wfrag_gk  = (unsigned short*)(ws + 2 * lsz + 16384);         // 24 KB
    unsigned short* wfrag_grk = (unsigned short*)(ws + 2 * lsz + 16384 + 24576); // 24 KB
    float*          gsum      = (float*)(ws + 2 * lsz + 16384 + 2 * 24576);      // [64]

    long nstate = E * LSD;
    int cpgrid = (int)((nstate + 255) / 256 > 4096 ? 4096 : (nstate + 255) / 256);

    copy_f32_kernel<<<cpgrid, 256, 0, stream>>>(ls, link_state, nstate);
    prep_frag_kernel<<<32, 256, 0, stream>>>(W_msg, wfrag_msg, TWOD, LSD);
    prep_frag_kernel<<<48, 256, 0, stream>>>(gk,    wfrag_gk,  LSD,  192);
    prep_frag_kernel<<<48, 256, 0, stream>>>(grk,   wfrag_grk, LSD,  192);
    zero_f32_kernel<<<1, 256, 0, stream>>>(gsum, 64);

    int msg_grid = (int)((P + 31) / 32);
    int gru_grid = (int)((E + 31) / 32);

    for (int step = 0; step < 4; ++step) {
        zero_f32_kernel<<<cpgrid, 256, 0, stream>>>(agg, nstate);
        msg_scatter_kernel<<<msg_grid, 256, 0, stream>>>(ls, first, second,
                                                         wfrag_msg, b_msg, agg, P);
        gru_wmma_kernel<<<gru_grid, 256, 0, stream>>>(agg, ls, wfrag_gk, wfrag_grk,
                                                      gb, E);
    }

    reduce_kernel<<<512, 256, 0, stream>>>(ls, gsum, E);
    mlp_kernel<<<1, 256, 0, stream>>>(gsum, W1, b1, W2, b2, W3, b3, (float*)d_out);
}
